// TransitionDown_72567767433470
// MI455X (gfx1250) — matrix operand order repaired and checked by reading, not compiled, not run on hardware
//
#include <hip/hip_runtime.h>

typedef __attribute__((ext_vector_type(16))) _Float16 v16h;
typedef __attribute__((ext_vector_type(8)))  float    v8f;

#define NPTS 16384
#define NS   16

union AFrag { v16h v; unsigned int u[8]; };

// ---------------------------------------------------------------------------
// Pack a row-major f32 B matrix [256 x NC] into per-lane WMMA B-fragment
// layout (f16):  frag(nt, kt) at ((nt*8 + kt)*32 + lane)*16 halves.
// lane 0-15: N = nt*16+lane, halves h -> K = kt*32 + h
// lane 16-31: N = nt*16+lane-16, halves h -> K = kt*32 + 16 + h
// ---------------------------------------------------------------------------
__global__ __launch_bounds__(256) void pack_b_kernel(
    const float* __restrict__ B, _Float16* __restrict__ P, int NC)
{
    int t = blockIdx.x * blockDim.x + threadIdx.x;
    int lane = t & 31;
    int kt   = (t >> 5) & 7;
    int nt   = t >> 8;
    if (nt >= NC / 16) return;
    int ncol  = nt * 16 + (lane & 15);
    int kbase = kt * 32 + (lane >> 4) * 16;
    _Float16* dst = P + ((size_t)(nt * 8 + kt) * 32 + lane) * 16;
#pragma unroll
    for (int h = 0; h < 16; ++h)
        dst[h] = (_Float16)B[(size_t)(kbase + h) * NC + ncol];
}

__global__ __launch_bounds__(256) void f32_to_f16_kernel(
    const float* __restrict__ x, _Float16* __restrict__ y, int n)
{
    int i = blockIdx.x * blockDim.x + threadIdx.x;
    if (i < n) y[i] = (_Float16)x[i];
}

// ---------------------------------------------------------------------------
// Brute-force KNN (top-16, includes self).  One thread per query point,
// candidates tiled through LDS.  Top-16 kept fully in registers via
// branchless compare-exchange insertion.
// ---------------------------------------------------------------------------
__global__ __launch_bounds__(256) void knn_kernel(
    const float* __restrict__ p, int* __restrict__ gidx, int N)
{
    __shared__ float tpx[256], tpy[256], tpz[256];
    int q = blockIdx.x * 256 + threadIdx.x;
    float qx = p[q * 3 + 0], qy = p[q * 3 + 1], qz = p[q * 3 + 2];
    float dist[16]; int idxk[16];
#pragma unroll
    for (int i = 0; i < 16; ++i) { dist[i] = 3.4e38f; idxk[i] = 0; }

    for (int base = 0; base < N; base += 256) {
        __syncthreads();
        int c = base + threadIdx.x;
        tpx[threadIdx.x] = p[c * 3 + 0];
        tpy[threadIdx.x] = p[c * 3 + 1];
        tpz[threadIdx.x] = p[c * 3 + 2];
        __syncthreads();
        for (int j = 0; j < 256; ++j) {
            float dx = qx - tpx[j], dy = qy - tpy[j], dz = qz - tpz[j];
            float d2 = dx * dx + dy * dy + dz * dz;
            if (d2 < dist[15]) {
                dist[15] = d2; idxk[15] = base + j;
#pragma unroll
                for (int s = 15; s > 0; --s) {
                    bool sw = dist[s] < dist[s - 1];
                    float td = dist[s]; int ti = idxk[s];
                    float tu = dist[s - 1]; int tj = idxk[s - 1];
                    dist[s]     = sw ? tu : td;  idxk[s]     = sw ? tj : ti;
                    dist[s - 1] = sw ? td : tu;  idxk[s - 1] = sw ? ti : tj;
                }
            }
        }
    }
#pragma unroll
    for (int i = 0; i < 16; ++i) gidx[q * 16 + i] = idxk[i];
}

// ---------------------------------------------------------------------------
// Generic K=256 GEMM:  out[M x NC] = A[M x 256](f16) @ Bpack + bias
// 8 waves/block, each wave owns a 16-row M-stripe; A-frags for all 8 k-steps
// preloaded into registers and reused across every N-tile.
// ---------------------------------------------------------------------------
template <int OUTF32>
__global__ __launch_bounds__(256) void gemm256_kernel(
    const _Float16* __restrict__ A, const _Float16* __restrict__ Bpack,
    const float* __restrict__ bias, void* __restrict__ out, int NC)
{
    const int lane = threadIdx.x & 31;
    const int wave = threadIdx.x >> 5;
    const int mbase = blockIdx.x * 128 + wave * 16;
    const int lrow = lane & 15;
    const int kblk = lane >> 4;

    AFrag a[8];
    const _Float16* arow = A + (size_t)(mbase + lrow) * 256;
#pragma unroll
    for (int kt = 0; kt < 8; ++kt)
#pragma unroll
        for (int i = 0; i < 8; ++i) {
            int k0 = kt * 32 + 16 * kblk + 2 * i;
            a[kt].u[i] = *(const unsigned int*)(arow + k0);
        }

    const v16h* bp = (const v16h*)Bpack;
    for (int nt = 0; nt < NC / 16; ++nt) {
        float bv = bias ? bias[nt * 16 + lrow] : 0.0f;
        v8f c;
#pragma unroll
        for (int v = 0; v < 8; ++v) c[v] = bv;
#pragma unroll
        for (int kt = 0; kt < 8; ++kt) {
            v16h b = bp[(size_t)(nt * 8 + kt) * 32 + lane];
            c = __builtin_amdgcn_wmma_f32_16x16x32_f16(
                    false, a[kt].v, false, b, (short)0, c, false, false);
        }
#pragma unroll
        for (int v = 0; v < 8; ++v) {
            int row = mbase + v + 8 * kblk;
            int col = nt * 16 + lrow;
            if (OUTF32)
                ((float*)out)[(size_t)row * NC + col] = c[v];
            else
                ((_Float16*)out)[(size_t)row * NC + col] = (_Float16)c[v];
        }
    }
}

// ---------------------------------------------------------------------------
// Fused PPF + positional-encoding GEMM + local attention.
// One block = 4 query points = 64 (query,neighbor) pair rows.
//   ph1: ppf features + hidden = relu(ppf@W1+b1) -> LDS (f16)
//   ph2: gather K/V rows (f16) into LDS, stage q rows
//   ph3: pe = hidden @ W2 + b2 via WMMA; epilogue adds pe into K and V in LDS
//   ph4: per-(query,head) attention with shuffle-based softmax
// ---------------------------------------------------------------------------
__global__ __launch_bounds__(256) void attn_kernel(
    const float* __restrict__ p, const float* __restrict__ nrm,
    const int* __restrict__ gidx, const _Float16* __restrict__ qkv,
    const float* __restrict__ W1, const float* __restrict__ b1,
    const _Float16* __restrict__ W2pack, const float* __restrict__ b2,
    _Float16* __restrict__ attnout)
{
    extern __shared__ _Float16 smem[];
    _Float16* hid = smem;               // 64 x 256
    _Float16* kh  = smem + 64 * 256;    // 64 x 256
    _Float16* vh  = smem + 2 * 64 * 256;

    __shared__ _Float16 qbuf[4 * 256];
    __shared__ float ppfbuf[64 * 4];
    __shared__ int   gbuf[64];
    __shared__ float w1buf[4 * 256];
    __shared__ float b1buf[256];
    __shared__ float b2buf[256];

    const int tid = threadIdx.x;
    const int qbase = blockIdx.x * 4;

    // stage small weights
    b1buf[tid] = b1[tid];
    b2buf[tid] = b2[tid];
#pragma unroll
    for (int i = 0; i < 4; ++i) w1buf[i * 256 + tid] = W1[i * 256 + tid];

    // --- phase 1a: point-pair features (threads 0..63, one per pair) ---
    if (tid < 64) {
        int m = qbase + (tid >> 4);
        int j = tid & 15;
        int g = gidx[m * 16 + j];
        gbuf[tid] = g;
        float px = p[m * 3], py = p[m * 3 + 1], pz = p[m * 3 + 2];
        float nx = nrm[m * 3], ny = nrm[m * 3 + 1], nz = nrm[m * 3 + 2];
        float gx = p[g * 3], gy = p[g * 3 + 1], gz = p[g * 3 + 2];
        float cx = nrm[g * 3], cy = nrm[g * 3 + 1], cz = nrm[g * 3 + 2];
        float vx = gx - px, vy = gy - py, vz = gz - pz;
        float d = sqrtf(vx * vx + vy * vy + vz * vz);
        auto ang = [](float ux, float uy, float uz,
                      float wx, float wy, float wz) {
            float rx = uy * wz - uz * wy;
            float ry = uz * wx - ux * wz;
            float rz = ux * wy - uy * wx;
            float cr = sqrtf(rx * rx + ry * ry + rz * rz);
            float dt = ux * wx + uy * wy + uz * wz;
            return atan2f(cr, dt);
        };
        ppfbuf[tid * 4 + 0] = ang(nx, ny, nz, vx, vy, vz);  // angle(nr, vec)
        ppfbuf[tid * 4 + 1] = ang(cx, cy, cz, vx, vy, vz);  // angle(c_n, vec)
        ppfbuf[tid * 4 + 2] = ang(nx, ny, nz, cx, cy, cz);  // angle(nr, c_n)
        ppfbuf[tid * 4 + 3] = d;
    }
    __syncthreads();

    // --- phase 1b: hidden = relu(ppf @ W1 + b1) -> f16 LDS ---
    {
        int pr = tid >> 2, seg = tid & 3;
        float f0 = ppfbuf[pr * 4 + 0], f1 = ppfbuf[pr * 4 + 1];
        float f2 = ppfbuf[pr * 4 + 2], f3 = ppfbuf[pr * 4 + 3];
        int hb = seg * 64;
        for (int h = 0; h < 64; ++h) {
            int hc = hb + h;
            float acc = b1buf[hc] + f0 * w1buf[hc] + f1 * w1buf[256 + hc] +
                        f2 * w1buf[512 + hc] + f3 * w1buf[768 + hc];
            hid[pr * 256 + hc] = (_Float16)fmaxf(acc, 0.0f);
        }
    }
    // --- phase 2: gather K,V neighbor rows; stage q rows ---
    {
        int pr = tid >> 2, seg = tid & 3;
        int g = gbuf[pr];
        const uint4* ksrc = (const uint4*)(qkv + (size_t)g * 768 + 256 + seg * 64);
        const uint4* vsrc = (const uint4*)(qkv + (size_t)g * 768 + 512 + seg * 64);
        uint4* kdst = (uint4*)(kh + pr * 256 + seg * 64);
        uint4* vdst = (uint4*)(vh + pr * 256 + seg * 64);
#pragma unroll
        for (int i = 0; i < 8; ++i) { kdst[i] = ksrc[i]; vdst[i] = vsrc[i]; }

        int m = tid >> 6, off = (tid & 63) * 4;
        *(uint2*)(qbuf + m * 256 + off) =
            *(const uint2*)(qkv + (size_t)(qbase + m) * 768 + off);
    }
    __syncthreads();

    // --- phase 3: pe GEMM (WMMA) with epilogue adding into kh, vh ---
    {
        int wave = tid >> 5, lane = tid & 31;
        int mtile = wave >> 1;
        int ntbase = (wave & 1) * 8;
        int lrow = lane & 15, kblk = lane >> 4;

        AFrag a[8];
        const _Float16* arow = hid + (mtile * 16 + lrow) * 256;
#pragma unroll
        for (int kt = 0; kt < 8; ++kt)
#pragma unroll
            for (int i = 0; i < 8; ++i) {
                int k0 = kt * 32 + 16 * kblk + 2 * i;
                a[kt].u[i] = *(const unsigned int*)(arow + k0);
            }
        const v16h* bp = (const v16h*)W2pack;
        for (int nn = 0; nn < 8; ++nn) {
            int nt = ntbase + nn;
            float bv = b2buf[nt * 16 + lrow];
            v8f c;
#pragma unroll
            for (int v = 0; v < 8; ++v) c[v] = bv;
#pragma unroll
            for (int kt = 0; kt < 8; ++kt) {
                v16h b = bp[(size_t)(nt * 8 + kt) * 32 + lane];
                c = __builtin_amdgcn_wmma_f32_16x16x32_f16(
                        false, a[kt].v, false, b, (short)0, c, false, false);
            }
#pragma unroll
            for (int v = 0; v < 8; ++v) {
                int row = mtile * 16 + v + 8 * kblk;
                int col = nt * 16 + lrow;
                int off = row * 256 + col;
                kh[off] = (_Float16)((float)kh[off] + c[v]);
                vh[off] = (_Float16)((float)vh[off] + c[v]);
            }
        }
    }
    __syncthreads();

    // --- phase 4: attention, one wave handles 2 (query,head) units ---
    {
        int wave = tid >> 5, lane = tid & 31;
#pragma unroll
        for (int u = 0; u < 2; ++u) {
            int unit = wave * 2 + u;
            int m = unit >> 2, h = unit & 3;
            int j = lane & 15, half = lane >> 4;
            const _Float16* kr = kh + (m * 16 + j) * 256 + h * 64 + half * 32;
            const _Float16* qr = qbuf + m * 256 + h * 64 + half * 32;
            float s = 0.f;
#pragma unroll
            for (int d = 0; d < 32; ++d) s += (float)qr[d] * (float)kr[d];
            s += __shfl_xor(s, 16, 32);
            s *= 0.125f;  // 1/sqrt(dh=64)
            float mx = s;
            mx = fmaxf(mx, __shfl_xor(mx, 1, 32));
            mx = fmaxf(mx, __shfl_xor(mx, 2, 32));
            mx = fmaxf(mx, __shfl_xor(mx, 4, 32));
            mx = fmaxf(mx, __shfl_xor(mx, 8, 32));
            float e = __expf(s - mx);
            float sum = e;
            sum += __shfl_xor(sum, 1, 32);
            sum += __shfl_xor(sum, 2, 32);
            sum += __shfl_xor(sum, 4, 32);
            sum += __shfl_xor(sum, 8, 32);
            float aat = e / sum;

            float acc0 = 0.f, acc1 = 0.f;
#pragma unroll
            for (int j2 = 0; j2 < 16; ++j2) {
                float aj = __shfl(aat, j2, 32);
                const _Float16* vr = vh + (m * 16 + j2) * 256 + h * 64;
                acc0 += aj * (float)vr[lane];
                acc1 += aj * (float)vr[32 + lane];
            }
            attnout[(size_t)(qbase + m) * 256 + h * 64 + lane]      = (_Float16)acc0;
            attnout[(size_t)(qbase + m) * 256 + h * 64 + 32 + lane] = (_Float16)acc1;
        }
    }
}

// ---------------------------------------------------------------------------
// Workspace layout (bytes)
// ---------------------------------------------------------------------------
static const size_t G_OFF    = 0;                            // gidx: N*16 int
static const size_t XH_OFF   = G_OFF + (size_t)NPTS * 16 * 4;
static const size_t QKV_OFF  = XH_OFF + (size_t)NPTS * 256 * 2;
static const size_t AO_OFF   = QKV_OFF + (size_t)NPTS * 768 * 2;
static const size_t WQKV_OFF = AO_OFF + (size_t)NPTS * 256 * 2;
static const size_t W2_OFF   = WQKV_OFF + (size_t)3 * 65536 * 2;
static const size_t WO_OFF   = W2_OFF + (size_t)65536 * 2;

extern "C" void kernel_launch(void* const* d_in, const int* in_sizes, int n_in,
                              void* d_out, int out_size, void* d_ws, size_t ws_size,
                              hipStream_t stream)
{
    const float* p   = (const float*)d_in[0];
    const float* x   = (const float*)d_in[1];
    const float* nrm = (const float*)d_in[3];
    const float* Wq  = (const float*)d_in[5];
    const float* Wk  = (const float*)d_in[6];
    const float* Wv  = (const float*)d_in[7];
    const float* W1  = (const float*)d_in[8];
    const float* b1  = (const float*)d_in[9];
    const float* W2  = (const float*)d_in[10];
    const float* b2  = (const float*)d_in[11];
    const float* Wo  = (const float*)d_in[12];
    const float* bo  = (const float*)d_in[13];
    float* out = (float*)d_out;

    char* ws = (char*)d_ws;
    int*       gidx     = (int*)(ws + G_OFF);
    _Float16*  xh       = (_Float16*)(ws + XH_OFF);
    _Float16*  qkv      = (_Float16*)(ws + QKV_OFF);
    _Float16*  attnout  = (_Float16*)(ws + AO_OFF);
    _Float16*  wqkvpack = (_Float16*)(ws + WQKV_OFF);
    _Float16*  w2pack   = (_Float16*)(ws + W2_OFF);
    _Float16*  wopack   = (_Float16*)(ws + WO_OFF);

    // weight packing (f32 -> WMMA B-fragment f16)
    pack_b_kernel<<<16, 256, 0, stream>>>(Wq, wqkvpack + 0 * 65536, 256);
    pack_b_kernel<<<16, 256, 0, stream>>>(Wk, wqkvpack + 1 * 65536, 256);
    pack_b_kernel<<<16, 256, 0, stream>>>(Wv, wqkvpack + 2 * 65536, 256);
    pack_b_kernel<<<16, 256, 0, stream>>>(W2, w2pack, 256);
    pack_b_kernel<<<16, 256, 0, stream>>>(Wo, wopack, 256);

    // x -> f16
    f32_to_f16_kernel<<<(NPTS * 256 + 255) / 256, 256, 0, stream>>>(
        x, xh, NPTS * 256);

    // KNN
    knn_kernel<<<NPTS / 256, 256, 0, stream>>>(p, gidx, NPTS);

    // Q|K|V = x @ [Wq|Wk|Wv]   (three 256-col groups, stored interleaved 768)
    gemm256_kernel<0><<<NPTS / 128, 256, 0, stream>>>(
        xh, wqkvpack, nullptr, qkv, 768);

    // fused ppf + pe-GEMM + attention  (96 KB dynamic LDS for hid/kh/vh)
    attn_kernel<<<NPTS / 4, 256, 3 * 64 * 256 * sizeof(_Float16), stream>>>(
        p, nrm, gidx, qkv, W1, b1, w2pack, b2, attnout);

    // out = attnout @ Wo + bo  (f32)
    gemm256_kernel<1><<<NPTS / 128, 256, 0, stream>>>(
        attnout, wopack, bo, out, 256);
}
